// OFTModule_87540023427222
// MI455X (gfx1250) — matrix operand order repaired and checked by reading, not compiled, not run on hardware
//
#include <hip/hip_runtime.h>
#include <math.h>

// ---------------- types ----------------
typedef __attribute__((ext_vector_type(16))) __bf16 v16bf;
typedef __attribute__((ext_vector_type(8)))  float  v8f;
typedef __attribute__((ext_vector_type(4)))  float  v4f;
typedef __attribute__((ext_vector_type(8)))  unsigned short u16x8;
typedef __attribute__((ext_vector_type(4)))  unsigned short u16x4;

#define D   1280
#define NB  16
#define BS  80
#define BM  128
#define BN  128
#define BK  32
#define LDK 40   // padded LDS K-stride in halves (80 bytes = 20 banks, conflict-free b128)

static __device__ __forceinline__ unsigned short f2bf(float f) {
    unsigned int u = __float_as_uint(f);
    unsigned int r = (u + 0x7fffu + ((u >> 16) & 1u)) >> 16;   // round-to-nearest-even
    return (unsigned short)r;
}

// ---------------- 1) global norm of skew(Q), clamped scale ----------------
__global__ __launch_bounds__(256) void oft_norm_scale(const float* __restrict__ ob,
                                                      float* __restrict__ scale) {
    __shared__ float red[256];
    float ss = 0.f;
    for (int idx = threadIdx.x; idx < NB * BS * BS; idx += 256) {
        int n = idx / (BS * BS);
        int rem = idx - n * BS * BS;
        int r = rem / BS, c = rem - r * BS;
        float v = ob[n * BS * BS + r * BS + c] - ob[n * BS * BS + c * BS + r];
        ss += v * v;
    }
    red[threadIdx.x] = ss;
    __syncthreads();
    for (int s = 128; s > 0; s >>= 1) {
        if ((int)threadIdx.x < s) red[threadIdx.x] += red[threadIdx.x + s];
        __syncthreads();
    }
    if (threadIdx.x == 0) {
        float norm = sqrtf(red[0]);
        scale[0] = (fminf(norm, (float)D) + 1e-8f) / (norm + 1e-8f);
    }
}

// ---------------- 2) per-block Cayley: R = (I+Q)(I-Q)^-1, Gauss-Jordan in LDS ----------------
__global__ __launch_bounds__(256) void oft_cayley(const float* __restrict__ ob,
                                                  const float* __restrict__ scale,
                                                  float* __restrict__ R) {
    __shared__ float Ainv[BS * BS];   // starts as I-Q, becomes its inverse in place
    __shared__ float P[BS * BS];      // I+Q
    __shared__ float col[BS];
    __shared__ float pinv_s;
    const int n = blockIdx.x;
    const float s = scale[0];
    const float* blk = ob + n * BS * BS;

    for (int idx = threadIdx.x; idx < BS * BS; idx += 256) {
        int r = idx / BS, c = idx - r * BS;
        float q  = s * (blk[r * BS + c] - blk[c * BS + r]);
        float id = (r == c) ? 1.f : 0.f;
        Ainv[idx] = id - q;
        P[idx]    = id + q;
    }
    __syncthreads();

    for (int p = 0; p < BS; ++p) {
        if ((int)threadIdx.x < BS) col[threadIdx.x] = Ainv[threadIdx.x * BS + p];
        if (threadIdx.x == 0) pinv_s = 1.f / Ainv[p * BS + p];
        __syncthreads();
        float pinv = pinv_s;
        if ((int)threadIdx.x < BS) {
            int j = threadIdx.x;
            Ainv[p * BS + j] = (j == p) ? pinv : Ainv[p * BS + j] * pinv;
        }
        __syncthreads();
        for (int idx = threadIdx.x; idx < BS * BS; idx += 256) {
            int r = idx / BS, c = idx - r * BS;
            if (r != p) Ainv[idx] -= col[r] * Ainv[p * BS + c];
        }
        __syncthreads();
        if ((int)threadIdx.x < BS && (int)threadIdx.x != p) {
            Ainv[threadIdx.x * BS + p] = -col[threadIdx.x] * pinv;
        }
        __syncthreads();
    }

    for (int idx = threadIdx.x; idx < BS * BS; idx += 256) {
        int r = idx / BS, c = idx - r * BS;
        float acc = 0.f;
        #pragma unroll 8
        for (int k = 0; k < BS; ++k) acc += P[r * BS + k] * Ainv[k * BS + c];
        R[n * BS * BS + idx] = acc;
    }
}

// ---------------- 3) rotated weights: Wp[o', i] = sum_k W[n*80+k, i] * R[n,k,j], bf16 [N][K] ----------------
__global__ __launch_bounds__(256) void oft_build_wp(const float* __restrict__ W,
                                                    const float* __restrict__ R,
                                                    unsigned short* __restrict__ Wpb) {
    int e = blockIdx.x * 256 + threadIdx.x;    // D*D elements; each block = part of one row (1280 = 5*256)
    int o = e / D, i = e - o * D;
    int n = o / BS, j = o - n * BS;
    const float* wcol = W + (size_t)(n * BS) * D + i;
    const float* rr   = R + n * BS * BS + j;
    float acc = 0.f;
    #pragma unroll 8
    for (int k = 0; k < BS; ++k) acc += wcol[(size_t)k * D] * rr[k * BS];
    Wpb[(size_t)o * D + i] = f2bf(acc);
}

// ---------------- 4) rotated bias ----------------
__global__ __launch_bounds__(256) void oft_build_bp(const float* __restrict__ b,
                                                    const float* __restrict__ R,
                                                    float* __restrict__ bp) {
    int o = blockIdx.x * 256 + threadIdx.x;
    if (o >= D) return;
    int n = o / BS, j = o - n * BS;
    float acc = 0.f;
    #pragma unroll 8
    for (int k = 0; k < BS; ++k) acc += b[n * BS + k] * R[n * BS * BS + k * BS + j];
    bp[o] = acc;
}

// ---------------- 5) fused GEMM: out = x @ Wp^T + bp, bf16 WMMA, f32 accum, double-buffered LDS ----------------
__global__ __launch_bounds__(256) void oft_gemm(const float* __restrict__ x,
                                                const unsigned short* __restrict__ Wpb,
                                                const float* __restrict__ bp,
                                                float* __restrict__ out) {
    __shared__ unsigned short As[2][BM * LDK];   // [buf][row][k] bf16, 80B row stride
    __shared__ unsigned short Bs[2][BN * LDK];

    const int tid  = threadIdx.x;
    const int lane = tid & 31;
    const int wid  = tid >> 5;
    const int wm   = wid >> 1;          // 0..3 -> wave M offset = wm*32
    const int wn   = wid & 1;           // 0..1 -> wave N offset = wn*64
    const int m0   = blockIdx.y * BM;
    const int n0   = blockIdx.x * BN;
    const int l15  = lane & 15;
    const int lh   = lane >> 4;         // 0 or 1 (half-wave)

    // per-thread staging coordinates (constant over K loop)
    const int arow = tid >> 3, aseg = tid & 7;   // A: rows arow + i*32, 8 segs of 4 floats
    const int brow = tid >> 2, bseg = tid & 3;   // B: rows brow + i*64, 4 segs of 8 halves

    v8f acc[2][4];
    #pragma unroll
    for (int mt = 0; mt < 2; ++mt)
        #pragma unroll
        for (int nt = 0; nt < 4; ++nt) acc[mt][nt] = (v8f){};

    v4f   areg[4];
    u16x8 breg[2];

    auto load_tile = [&](int k0) {
        #pragma unroll
        for (int i = 0; i < 4; ++i)
            areg[i] = *(const v4f*)(x + (size_t)(m0 + arow + i * 32) * D + k0 + aseg * 4);
        #pragma unroll
        for (int i = 0; i < 2; ++i)
            breg[i] = *(const u16x8*)(Wpb + (size_t)(n0 + brow + i * 64) * D + k0 + bseg * 8);
    };
    auto store_tile = [&](int buf) {
        #pragma unroll
        for (int i = 0; i < 4; ++i) {
            u16x4 h;
            h[0] = f2bf(areg[i][0]); h[1] = f2bf(areg[i][1]);
            h[2] = f2bf(areg[i][2]); h[3] = f2bf(areg[i][3]);
            *(u16x4*)(&As[buf][(arow + i * 32) * LDK + aseg * 4]) = h;   // ds_store_b64
        }
        #pragma unroll
        for (int i = 0; i < 2; ++i) {
            union { u16x8 v; u16x4 h[2]; } u; u.v = breg[i];
            *(u16x4*)(&Bs[buf][(brow + i * 64) * LDK + bseg * 8])     = u.h[0];
            *(u16x4*)(&Bs[buf][(brow + i * 64) * LDK + bseg * 8 + 4]) = u.h[1];
        }
    };

    // prologue: stage tile 0
    load_tile(0);
    store_tile(0);
    int cur = 0;

    for (int k0 = 0; k0 < D; k0 += BK) {
        const bool more = (k0 + BK < D);        // uniform across block
        __syncthreads();                        // publish LDS[cur]

        if (more) load_tile(k0 + BK);           // global loads overlap WMMAs below
        if (k0 + 2 * BK < D)
            __builtin_prefetch(x + (size_t)(m0 + arow) * D + (k0 + 2 * BK) + aseg * 4, 0, 0);

        // ---- A fragments: 16x32 bf16 ISA layout (lane<16: K0-7/16-23; lane>=16: K8-15/24-31) ----
        v16bf afrag[2];
        #pragma unroll
        for (int mt = 0; mt < 2; ++mt) {
            int row = wm * 32 + mt * 16 + l15;
            const unsigned short* base = &As[cur][row * LDK + lh * 8];
            union { v16bf v; u16x8 h[2]; } u;
            u.h[0] = *(const u16x8*)(base);        // K = lh*8 .. +8
            u.h[1] = *(const u16x8*)(base + 16);   // K = 16+lh*8 .. +8
            afrag[mt] = u.v;
        }
        // ---- B fragments: 32x16 bf16 (stored N-major), lane holds N=l15, K = lh*8..+8 and 16+lh*8..+8 ----
        v16bf bfrag[4];
        #pragma unroll
        for (int nt = 0; nt < 4; ++nt) {
            int row = wn * 64 + nt * 16 + l15;
            const unsigned short* base = &Bs[cur][row * LDK + lh * 8];
            union { v16bf v; u16x8 h[2]; } u;
            u.h[0] = *(const u16x8*)(base);
            u.h[1] = *(const u16x8*)(base + 16);
            bfrag[nt] = u.v;
        }

        // ---- 8 WMMAs: 32x64 wave tile ----
        #pragma unroll
        for (int mt = 0; mt < 2; ++mt)
            #pragma unroll
            for (int nt = 0; nt < 4; ++nt)
                acc[mt][nt] = __builtin_amdgcn_wmma_f32_16x16x32_bf16(
                    false, afrag[mt], false, bfrag[nt],
                    (short)0, acc[mt][nt], false, false);

        if (more) store_tile(cur ^ 1);          // fill other buffer; safe, nobody reads it yet
        cur ^= 1;
    }

    // ---- epilogue: C layout (VGPR r -> M = r + lh*8), add bias, store fp32 ----
    float bpv[4];
    #pragma unroll
    for (int nt = 0; nt < 4; ++nt) bpv[nt] = bp[n0 + wn * 64 + nt * 16 + l15];

    #pragma unroll
    for (int mt = 0; mt < 2; ++mt) {
        #pragma unroll
        for (int nt = 0; nt < 4; ++nt) {
            int col = n0 + wn * 64 + nt * 16 + l15;
            #pragma unroll
            for (int r = 0; r < 8; ++r) {
                int row = m0 + wm * 32 + mt * 16 + lh * 8 + r;
                out[(size_t)row * D + col] = acc[mt][nt][r] + bpv[nt];
            }
        }
    }
}

// ---------------- host launcher ----------------
extern "C" void kernel_launch(void* const* d_in, const int* in_sizes, int n_in,
                              void* d_out, int out_size, void* d_ws, size_t ws_size,
                              hipStream_t stream) {
    const float* x  = (const float*)d_in[0];   // [4,8192,1280]
    const float* W  = (const float*)d_in[1];   // [1280,1280]
    const float* b  = (const float*)d_in[2];   // [1280]
    const float* ob = (const float*)d_in[3];   // [16,80,80]
    float* out = (float*)d_out;
    const int M = in_sizes[0] / D;             // 32768

    // workspace: scale(256B) | R fp32 (400KB) | Wp bf16 (3.125MB) | bp fp32 (5KB)
    char* ws = (char*)d_ws;
    float* scale        = (float*)ws;
    float* R            = (float*)(ws + 256);
    unsigned short* Wpb = (unsigned short*)(ws + 256 + (size_t)NB * BS * BS * 4);
    float* bp           = (float*)(ws + 256 + (size_t)NB * BS * BS * 4 + (size_t)D * D * 2);

    oft_norm_scale<<<1, 256, 0, stream>>>(ob, scale);
    oft_cayley<<<NB, 256, 0, stream>>>(ob, scale, R);
    oft_build_wp<<<(D * D) / 256, 256, 0, stream>>>(W, R, Wpb);
    oft_build_bp<<<(D + 255) / 256, 256, 0, stream>>>(b, R, bp);

    dim3 grid(D / BN, M / BM);
    oft_gemm<<<grid, 256, 0, stream>>>(x, Wpb, bp, out);
}
